// SupMemCrossAttention_45612552684056
// MI455X (gfx1250) — compile-verified, matrix operand order checked by reading, not run on hardware
//
#include <hip/hip_runtime.h>

typedef __attribute__((ext_vector_type(16))) __bf16 v16bf;
typedef __attribute__((ext_vector_type(8)))  float  v8f;

#define WMMA_BF16(a,b,c) \
  __builtin_amdgcn_wmma_f32_16x16x32_bf16(false,(a),false,(b),(short)0,(c),false,false)

#define DIMK 512
#define NQ   1024
#define NS   4096
#define NM   4096
#define NB   8
#define SCALE_F 0.044194173824159216f
#define NEG_F  (-10000.0f)

#define KSTRIDE 520   // 32 x 520 bf16 K chunk view (padded: 16 lanes hit distinct banks)
#define VSTRIDE 40    // 512 x 40 bf16 V chunk view (padded)

__device__ __forceinline__ v8f v8f_zero() {
  v8f z;
#pragma unroll
  for (int i = 0; i < 8; i++) z[i] = 0.0f;
  return z;
}

// 32-byte fragment load with only 16B alignment guaranteed (vmem/ds max is b128)
__device__ __forceinline__ v16bf ld2x16(const __bf16* p) {
  union { uint4 u[2]; v16bf v; } x;
  x.u[0] = *(const uint4*)p;
  x.u[1] = *(const uint4*)(p + 8);
  return x.v;
}

__device__ __forceinline__ void cvt16_store(__bf16* d, float4 f0, float4 f1,
                                            float4 f2, float4 f3) {
  d[0]=(__bf16)f0.x; d[1]=(__bf16)f0.y; d[2]=(__bf16)f0.z; d[3]=(__bf16)f0.w;
  d[4]=(__bf16)f1.x; d[5]=(__bf16)f1.y; d[6]=(__bf16)f1.z; d[7]=(__bf16)f1.w;
  d[8]=(__bf16)f2.x; d[9]=(__bf16)f2.y; d[10]=(__bf16)f2.z; d[11]=(__bf16)f2.w;
  d[12]=(__bf16)f3.x; d[13]=(__bf16)f3.y; d[14]=(__bf16)f3.z; d[15]=(__bf16)f3.w;
}

// ---------------------------------------------------------------------------
// GEMM: Out[M,512] = A[M,512] @ W[512,512]^T
// OUT_MODE: 0 = bf16 row-major, 1 = bf16 transposed per batch (d-major V),
//           2 = f32 row-major
// 64x64 block tile, BK=32, 4 waves, 2x2 WMMA frags; next K-tile prefetched
// into registers so global loads overlap the WMMA phase.
// ---------------------------------------------------------------------------
template<bool A_BF16, int OUT_MODE>
__global__ __launch_bounds__(128) void gemm_wt_kernel(
    const void* __restrict__ Ain, const float* __restrict__ W,
    void* __restrict__ Out, int batchRows)
{
  __shared__ __attribute__((aligned(64))) __bf16 As[64][48];
  __shared__ __attribute__((aligned(64))) __bf16 Bs[64][48];

  const int tid   = threadIdx.x;
  const int lane  = tid & 31;
  const int wave  = tid >> 5;
  const int wm    = (wave >> 1) * 32;
  const int wn    = (wave & 1) * 32;
  const int mBase = blockIdx.x * 64;
  const int nBase = blockIdx.y * 64;
  const int srow  = tid >> 1;
  const int shalf = (tid & 1) * 16;

  const __bf16* aRowB = (const __bf16*)Ain + ((size_t)(mBase + srow)) * DIMK + shalf;
  const float*  aRowF = (const float*)Ain + ((size_t)(mBase + srow)) * DIMK + shalf;
  const float*  wRow  = W + ((size_t)(nBase + srow)) * DIMK + shalf;

  v8f acc[2][2];
#pragma unroll
  for (int i = 0; i < 2; i++)
#pragma unroll
    for (int j = 0; j < 2; j++) acc[i][j] = v8f_zero();

  uint4  abuf0, abuf1;
  float4 af0, af1, af2, af3;
  float4 wf0, wf1, wf2, wf3;
  if constexpr (A_BF16) {
    const uint4* s = (const uint4*)aRowB;
    abuf0 = s[0]; abuf1 = s[1];
  } else {
    const float4* s = (const float4*)aRowF;
    af0 = s[0]; af1 = s[1]; af2 = s[2]; af3 = s[3];
  }
  { const float4* s = (const float4*)wRow;
    wf0 = s[0]; wf1 = s[1]; wf2 = s[2]; wf3 = s[3]; }

  for (int kk = 0; kk < DIMK; kk += 32) {
    if constexpr (A_BF16) {
      *(uint4*)&As[srow][shalf]     = abuf0;
      *(uint4*)&As[srow][shalf + 8] = abuf1;
    } else {
      cvt16_store(&As[srow][shalf], af0, af1, af2, af3);
    }
    cvt16_store(&Bs[srow][shalf], wf0, wf1, wf2, wf3);
    __syncthreads();

    const int kn = (kk + 32 < DIMK) ? kk + 32 : kk;
    if constexpr (A_BF16) {
      const uint4* s = (const uint4*)(aRowB + kn);
      abuf0 = s[0]; abuf1 = s[1];
    } else {
      const float4* s = (const float4*)(aRowF + kn);
      af0 = s[0]; af1 = s[1]; af2 = s[2]; af3 = s[3];
    }
    { const float4* s = (const float4*)(wRow + kn);
      wf0 = s[0]; wf1 = s[1]; wf2 = s[2]; wf3 = s[3]; }

    v16bf a0 = *(const v16bf*)&As[wm      + (lane & 15)][(lane >> 4) * 16];
    v16bf a1 = *(const v16bf*)&As[wm + 16 + (lane & 15)][(lane >> 4) * 16];
    v16bf b0 = *(const v16bf*)&Bs[wn      + (lane & 15)][(lane >> 4) * 16];
    v16bf b1 = *(const v16bf*)&Bs[wn + 16 + (lane & 15)][(lane >> 4) * 16];
    acc[0][0] = WMMA_BF16(a0, b0, acc[0][0]);
    acc[0][1] = WMMA_BF16(a0, b1, acc[0][1]);
    acc[1][0] = WMMA_BF16(a1, b0, acc[1][0]);
    acc[1][1] = WMMA_BF16(a1, b1, acc[1][1]);
    __syncthreads();
  }

#pragma unroll
  for (int i = 0; i < 2; i++)
#pragma unroll
    for (int j = 0; j < 2; j++)
#pragma unroll
      for (int r = 0; r < 8; r++) {
        int grow = mBase + wm + i * 16 + r + ((lane >> 4) << 3);
        int gcol = nBase + wn + j * 16 + (lane & 15);
        float v = acc[i][j][r];
        if constexpr (OUT_MODE == 0) {
          ((__bf16*)Out)[(size_t)grow * DIMK + gcol] = (__bf16)v;
        } else if constexpr (OUT_MODE == 1) {
          int bb  = grow / batchRows;
          int key = grow - bb * batchRows;
          ((__bf16*)Out)[((size_t)bb * DIMK + gcol) * (size_t)batchRows + key] = (__bf16)v;
        } else {
          ((float*)Out)[(size_t)grow * DIMK + gcol] = v;
        }
      }
}

// ---------------------------------------------------------------------------
// Flash cross-attention, joint softmax over s then m keys.
// 4 waves / WG; each wave owns 16 query rows (Q held in 16 v16bf registers).
// K and V chunks (32 keys) are staged cooperatively into ONE shared LDS
// buffer and consumed by all 4 waves: 4x higher FLOP/byte vs per-wave global
// streaming (the previous version pulled ~12 GB from L2 for ~200 GFLOP).
// Inner-loop operands become ds_load_b128 (dscnt, ~tens of cycles) instead of
// global loads (loadcnt, ~hundreds), removing the s_wait_loadcnt 0 stalls.
// Pass 1: joint running max/denominator. Pass 2: normalized P @ V with one
// 256-VGPR accumulator, flushed at the s/m boundary.
// ---------------------------------------------------------------------------
__global__ __launch_bounds__(128) void flash_kernel(
    const __bf16* __restrict__ q,
    const __bf16* __restrict__ sk, const __bf16* __restrict__ svt,
    const __bf16* __restrict__ mk, const __bf16* __restrict__ mvt,
    const int* __restrict__ smask, const int* __restrict__ mmask,
    __bf16* __restrict__ satt, __bf16* __restrict__ matt)
{
  // K view: 32 rows x KSTRIDE ; V view: 512 rows x VSTRIDE  (union, 40 KB)
  __shared__ __attribute__((aligned(64))) __bf16 kvsh[512 * VSTRIDE];
  __shared__ __attribute__((aligned(64))) __bf16 ps[4][16][32];     // 4 KB

  const int tid   = threadIdx.x;
  const int lane  = tid & 31;
  const int wave  = tid >> 5;
  const int b     = blockIdx.y;
  const int qBase = blockIdx.x * 64 + wave * 16;
  const int hoff  = (lane >> 4) << 4;   // 0 / 16 : K-half select
  const int l15   = lane & 15;

  // ---- Q tile in registers: qf[i] = A-frag for d = i*32 ----
  v16bf qf[16];
  {
    const __bf16* qrow = q + ((size_t)b * NQ + qBase + l15) * DIMK + hoff;
#pragma unroll
    for (int i = 0; i < 16; i++) qf[i] = ld2x16(qrow + i * 32);
  }

  float mrun[8], lrun[8];
#pragma unroll
  for (int r = 0; r < 8; r++) { mrun[r] = -1e30f; lrun[r] = 0.0f; }

  // ================= pass 1: joint max / denominator =================
  for (int kc = 0; kc < NS + NM; kc += 32) {
    const __bf16* kb; const int* msk; int key0;
    if (kc < NS) { kb = sk + (size_t)b * NS * DIMK; msk = smask + (size_t)b * NS; key0 = kc; }
    else         { kb = mk + (size_t)b * NM * DIMK; msk = mmask + (size_t)b * NM; key0 = kc - NS; }
    { int pk = key0 + 32 + (tid & 31);
      if (pk < NS) __builtin_prefetch(&kb[(size_t)pk * DIMK], 0, 1); }

    __syncthreads();                       // prior chunk reads complete
    {                                      // stage 32x512 K chunk (128 thr)
      int row = tid >> 2;
      int c0  = (tid & 3) * 128;
      const __bf16* g = kb + ((size_t)(key0 + row)) * DIMK + c0;
      __bf16* l = &kvsh[row * KSTRIDE + c0];
#pragma unroll
      for (int u = 0; u < 16; u++)
        *(uint4*)(l + u * 8) = *(const uint4*)(g + u * 8);
    }
    __syncthreads();

    v8f s0 = v8f_zero(), s1 = v8f_zero();
#pragma unroll
    for (int i = 0; i < 16; i++) {
      const __bf16* kp = &kvsh[l15 * KSTRIDE + i * 32 + hoff];
      v16bf b0 = ld2x16(kp);
      v16bf b1 = ld2x16(kp + 16 * KSTRIDE);
      s0 = WMMA_BF16(qf[i], b0, s0);
      s1 = WMMA_BF16(qf[i], b1, s1);
    }
    float bias0 = msk[key0 +      l15] ? 0.0f : NEG_F;
    float bias1 = msk[key0 + 16 + l15] ? 0.0f : NEG_F;
#pragma unroll
    for (int r = 0; r < 8; r++) {
      float x0 = s0[r] * SCALE_F + bias0;
      float x1 = s1[r] * SCALE_F + bias1;
      float mx = fmaxf(x0, x1);
#pragma unroll
      for (int off = 8; off >= 1; off >>= 1) mx = fmaxf(mx, __shfl_xor(mx, off, 32));
      float nm = fmaxf(mrun[r], mx);
      float p0 = __expf(x0 - nm), p1 = __expf(x1 - nm);
      float rs = p0 + p1;
#pragma unroll
      for (int off = 8; off >= 1; off >>= 1) rs += __shfl_xor(rs, off, 32);
      lrun[r] = lrun[r] * __expf(mrun[r] - nm) + rs;
      mrun[r] = nm;
    }
  }
  float invl[8];
#pragma unroll
  for (int r = 0; r < 8; r++) invl[r] = 1.0f / lrun[r];

  // ================= pass 2: P @ V, flush at s/m boundary =================
  v8f acc[32];
#pragma unroll
  for (int j = 0; j < 32; j++) acc[j] = v8f_zero();

  for (int kc = 0; kc < NS + NM; kc += 32) {
    const __bf16* kb; const __bf16* vtb; const int* msk; int key0; int nk;
    if (kc < NS) { kb = sk + (size_t)b * NS * DIMK; vtb = svt + (size_t)b * DIMK * NS;
                   msk = smask + (size_t)b * NS; key0 = kc;      nk = NS; }
    else         { kb = mk + (size_t)b * NM * DIMK; vtb = mvt + (size_t)b * DIMK * NM;
                   msk = mmask + (size_t)b * NM; key0 = kc - NS; nk = NM; }
    { int pk = key0 + 32 + (tid & 31);
      if (pk < nk) __builtin_prefetch(&kb[(size_t)pk * DIMK], 0, 1); }

    __syncthreads();                       // prior V reads complete
    {                                      // stage K chunk
      int row = tid >> 2;
      int c0  = (tid & 3) * 128;
      const __bf16* g = kb + ((size_t)(key0 + row)) * DIMK + c0;
      __bf16* l = &kvsh[row * KSTRIDE + c0];
#pragma unroll
      for (int u = 0; u < 16; u++)
        *(uint4*)(l + u * 8) = *(const uint4*)(g + u * 8);
    }
    __syncthreads();

    v8f s0 = v8f_zero(), s1 = v8f_zero();
#pragma unroll
    for (int i = 0; i < 16; i++) {
      const __bf16* kp = &kvsh[l15 * KSTRIDE + i * 32 + hoff];
      v16bf b0 = ld2x16(kp);
      v16bf b1 = ld2x16(kp + 16 * KSTRIDE);
      s0 = WMMA_BF16(qf[i], b0, s0);
      s1 = WMMA_BF16(qf[i], b1, s1);
    }

    __syncthreads();                       // K reads complete
    {                                      // stage 512x32 V^T chunk (d-major)
      int r0 = tid * 4;
#pragma unroll
      for (int rr = 0; rr < 4; rr++) {
        const __bf16* g = vtb + (size_t)(r0 + rr) * (size_t)nk + key0;
        __bf16* l = &kvsh[(r0 + rr) * VSTRIDE];
        *(uint4*)(l)      = *(const uint4*)(g);
        *(uint4*)(l + 8)  = *(const uint4*)(g + 8);
        *(uint4*)(l + 16) = *(const uint4*)(g + 16);
        *(uint4*)(l + 24) = *(const uint4*)(g + 24);
      }
    }
    __syncthreads();

    float bias0 = msk[key0 +      l15] ? 0.0f : NEG_F;
    float bias1 = msk[key0 + 16 + l15] ? 0.0f : NEG_F;
    // normalized probabilities -> LDS transpose bounce (C-layout -> A-layout)
#pragma unroll
    for (int r = 0; r < 8; r++) {
      float x0 = s0[r] * SCALE_F + bias0;
      float x1 = s1[r] * SCALE_F + bias1;
      float p0 = __expf(x0 - mrun[r]) * invl[r];
      float p1 = __expf(x1 - mrun[r]) * invl[r];
      int prow = r + ((lane >> 4) << 3);
      ps[wave][prow][l15]      = (__bf16)p0;
      ps[wave][prow][16 + l15] = (__bf16)p1;
    }
    v16bf pf = ld2x16(&ps[wave][l15][hoff]);
#pragma unroll
    for (int j = 0; j < 32; j++) {
      v16bf bv = ld2x16(&kvsh[(j * 16 + l15) * VSTRIDE + hoff]);
      acc[j] = WMMA_BF16(pf, bv, acc[j]);
    }

    if (kc == NS - 32) {  // flush s-attention output, reset accumulator
#pragma unroll
      for (int j = 0; j < 32; j++) {
#pragma unroll
        for (int r = 0; r < 8; r++) {
          int row = qBase + r + ((lane >> 4) << 3);
          satt[((size_t)b * NQ + row) * DIMK + j * 16 + l15] = (__bf16)acc[j][r];
        }
        acc[j] = v8f_zero();
      }
    }
  }
#pragma unroll
  for (int j = 0; j < 32; j++)
#pragma unroll
    for (int r = 0; r < 8; r++) {
      int row = qBase + r + ((lane >> 4) << 3);
      matt[((size_t)b * NQ + row) * DIMK + j * 16 + l15] = (__bf16)acc[j][r];
    }
}

// ---------------------------------------------------------------------------
extern "C" void kernel_launch(void* const* d_in, const int* in_sizes, int n_in,
                              void* d_out, int out_size, void* d_ws, size_t ws_size,
                              hipStream_t stream)
{
  (void)in_sizes; (void)n_in; (void)out_size; (void)ws_size;
  const float* prototype = (const float*)d_in[0];
  const float* s_x       = (const float*)d_in[1];
  const float* m_x       = (const float*)d_in[2];
  const int*   smask     = (const int*)d_in[3];
  const int*   mmask     = (const int*)d_in[4];
  const float* Wq        = (const float*)d_in[5];
  const float* Wk        = (const float*)d_in[6];
  const float* Wv        = (const float*)d_in[7];
  const float* Wp        = (const float*)d_in[8];

  size_t off = 0;
  auto take = [&](size_t bytes) -> void* {
    void* p = (char*)d_ws + off;
    off += (bytes + 255) & ~(size_t)255;
    return p;
  };
  __bf16* qb  = (__bf16*)take((size_t)NB * NQ * DIMK * 2);
  __bf16* skb = (__bf16*)take((size_t)NB * NS * DIMK * 2);
  __bf16* svt = (__bf16*)take((size_t)NB * NS * DIMK * 2);
  __bf16* mkb = (__bf16*)take((size_t)NB * NM * DIMK * 2);
  __bf16* mvt = (__bf16*)take((size_t)NB * NM * DIMK * 2);
  __bf16* sat = (__bf16*)take((size_t)NB * NQ * DIMK * 2);
  __bf16* mat = (__bf16*)take((size_t)NB * NQ * DIMK * 2);

  dim3 tb(128);
  // projections (f32 in -> bf16 out); V stored transposed (d-major) per batch
  gemm_wt_kernel<false, 0><<<dim3((NB * NQ) / 64, DIMK / 64), tb, 0, stream>>>(prototype, Wq, qb, 0);
  gemm_wt_kernel<false, 0><<<dim3((NB * NS) / 64, DIMK / 64), tb, 0, stream>>>(s_x, Wk, skb, 0);
  gemm_wt_kernel<false, 1><<<dim3((NB * NS) / 64, DIMK / 64), tb, 0, stream>>>(s_x, Wv, svt, NS);
  gemm_wt_kernel<false, 0><<<dim3((NB * NM) / 64, DIMK / 64), tb, 0, stream>>>(m_x, Wk, mkb, 0);
  gemm_wt_kernel<false, 1><<<dim3((NB * NM) / 64, DIMK / 64), tb, 0, stream>>>(m_x, Wv, mvt, NM);

  // fused joint-softmax cross attention -> s_att, m_att (bf16)
  flash_kernel<<<dim3(NQ / 64, NB), dim3(128), 0, stream>>>(
      qb, skb, svt, mkb, mvt, smask, mmask, sat, mat);

  // output projections (bf16 in -> f32 out), s_out then m_out
  float* outp = (float*)d_out;
  gemm_wt_kernel<true, 2><<<dim3((NB * NQ) / 64, DIMK / 64), tb, 0, stream>>>(sat, Wp, outp, 0);
  gemm_wt_kernel<true, 2><<<dim3((NB * NQ) / 64, DIMK / 64), tb, 0, stream>>>(
      mat, Wp, outp + (size_t)NB * NQ * DIMK, 0);
}
